// GroupedQueryAttention_75290776699189
// MI455X (gfx1250) — matrix operand order repaired and checked by reading, not compiled, not run on hardware
//
#include <hip/hip_runtime.h>
#include <hip/hip_bf16.h>
#include <math.h>

// ---------------- problem constants ----------------
#define DIMX      1024
#define NUM_HEADS 16
#define NUM_KV    4
#define HEAD_DIM  64
#define SEQ       2048
#define BATCH     2
#define NROWS     (BATCH * SEQ)        // 4096 total (b,l) rows
#define ATTN_SCALE 0.125f              // 64^-0.5
#define ROPE_C    0.2878231366242557f  // ln(10000)/32

typedef __bf16 bf16_t;
typedef __attribute__((ext_vector_type(16))) __bf16 v16bf;
typedef __attribute__((ext_vector_type(8)))  __bf16 v8bf;
typedef __attribute__((ext_vector_type(8)))  float  v8f;

// ---------------- helpers ----------------
static __device__ __forceinline__ bf16_t f2bf(float f) {
  unsigned int u = __builtin_bit_cast(unsigned int, f);
  unsigned int r = u + 0x7FFFu + ((u >> 16) & 1u);   // round-to-nearest-even
  unsigned short h = (unsigned short)(r >> 16);
  return __builtin_bit_cast(bf16_t, h);
}

static __device__ __forceinline__ v16bf cat8(v8bf lo, v8bf hi) {
  return __builtin_shufflevector(lo, hi, 0,1,2,3,4,5,6,7,8,9,10,11,12,13,14,15);
}

static __device__ __forceinline__ v8f wmma_bf16(v16bf a, v16bf b, v8f c) {
  // D = A(16x32 bf16) * B(32x16 bf16) + C(16x16 f32)
  return __builtin_amdgcn_wmma_f32_16x16x32_bf16(false, a, false, b, (short)0, c, false, false);
}

// A-operand loader: lane holds row m=lane&15; K chunk {16g..16g+15}
static __device__ __forceinline__ v16bf load_a32(const bf16_t* rowp, int g) {
  v8bf lo = *(const v8bf*)(rowp + 16 * g);
  v8bf hi = *(const v8bf*)(rowp + 16 * g + 8);
  return cat8(lo, hi);
}

// ---------------- kernel 0: f32 -> bf16 ----------------
__global__ void cvt_bf16_kernel(const float* __restrict__ src, bf16_t* __restrict__ dst, int n) {
  int i = blockIdx.x * blockDim.x + threadIdx.x;
  if (i < n) dst[i] = f2bf(src[i]);
}

// ---------------- kernel 1: transpose+convert W[K][N] -> Wt[N][K] bf16 ----------------
__global__ void transpose_cvt_kernel(const float* __restrict__ src, bf16_t* __restrict__ dst,
                                     int K, int N) {
  int i = blockIdx.x * blockDim.x + threadIdx.x;
  if (i >= K * N) return;
  int n = i % N, k = i / N;                 // coalesced read along N
  dst[(size_t)n * K + k] = f2bf(src[(size_t)k * N + n]);
}

// ---------------- kernel 2: fused QKV projection + RoPE ----------------
// Each wave computes a 16(rows of B*L) x 64(one head) output tile:
//  cg in [0,16)  -> Q head cg        (RoPE, store [B,H,L,64])
//  cg in [16,20) -> K head cg-16     (RoPE, store [B,KV,L,64])
//  cg in [20,24) -> V head cg-20     (store transposed [B,KV,64,L])
__global__ void __launch_bounds__(256)
qkv_rope_kernel(const bf16_t* __restrict__ xb,
                const bf16_t* __restrict__ wqT, const bf16_t* __restrict__ wkT,
                const bf16_t* __restrict__ wvT,
                bf16_t* __restrict__ qb, bf16_t* __restrict__ kb, bf16_t* __restrict__ vT) {
  const int lane = threadIdx.x & 31;
  const int wid  = blockIdx.x * (blockDim.x >> 5) + (threadIdx.x >> 5);
  const int cg   = wid % 24;
  const int rt   = wid / 24;
  const int row0 = rt * 16;
  const int nl   = lane & 15;
  const int g    = lane >> 4;

  const bf16_t* Wt;
  int seg, hloc;
  if (cg < 16)      { Wt = wqT; seg = 0; hloc = cg; }
  else if (cg < 20) { Wt = wkT; seg = 1; hloc = cg - 16; }
  else              { Wt = wvT; seg = 2; hloc = cg - 20; }

  const bf16_t* arow = xb + (size_t)(row0 + nl) * DIMX;
  const bf16_t* base = Wt + (size_t)(hloc * 64 + nl) * DIMX + 16 * g;

  v8f a0 = {}, a1 = {}, a2 = {}, a3 = {};
  for (int kk = 0; kk < DIMX; kk += 32) {
    v16bf a = load_a32(arow + kk, g);
    __builtin_prefetch(arow + kk + 256, 0, 0);
    v16bf b0 = *(const v16bf*)(base + kk);
    v16bf b1 = *(const v16bf*)(base + (size_t)16 * DIMX + kk);
    v16bf b2 = *(const v16bf*)(base + (size_t)32 * DIMX + kk);
    v16bf b3 = *(const v16bf*)(base + (size_t)48 * DIMX + kk);
    a0 = wmma_bf16(a, b0, a0);
    a1 = wmma_bf16(a, b1, a1);
    a2 = wmma_bf16(a, b2, a2);
    a3 = wmma_bf16(a, b3, a3);
  }

  if (seg != 2) {
    // RoPE: lane's columns are {nl, nl+16, nl+32, nl+48}; pair (d, d+32) => (a0,a2),(a1,a3)
    const float invf0 = __expf(-(float)nl * ROPE_C);
    const float invf1 = __expf(-(float)(nl + 16) * ROPE_C);
#pragma unroll
    for (int r = 0; r < 8; r++) {
      const int row = row0 + r + 8 * g;
      const int l   = row & (SEQ - 1);
      const int bi  = row >> 11;
      float s0, c0, s1, c1;
      sincosf((float)l * invf0, &s0, &c0);
      sincosf((float)l * invf1, &s1, &c1);
      float x0 = a0[r], x1 = a1[r], x2 = a2[r], x3 = a3[r];
      float y0 = x0 * c0 - x2 * s0;
      float y2 = x2 * c0 + x0 * s0;
      float y1 = x1 * c1 - x3 * s1;
      float y3 = x3 * c1 + x1 * s1;
      bf16_t* o = (seg == 0)
        ? qb + (((size_t)bi * NUM_HEADS + hloc) * SEQ + l) * HEAD_DIM
        : kb + (((size_t)bi * NUM_KV    + hloc) * SEQ + l) * HEAD_DIM;
      o[nl]      = f2bf(y0);
      o[nl + 16] = f2bf(y1);
      o[nl + 32] = f2bf(y2);
      o[nl + 48] = f2bf(y3);
    }
  } else {
    // V stored transposed: vT[b][kv][d][l]
#pragma unroll
    for (int r = 0; r < 8; r++) {
      const int row = row0 + r + 8 * g;
      const int l   = row & (SEQ - 1);
      const int bi  = row >> 11;
      bf16_t* vb = vT + ((size_t)bi * NUM_KV + hloc) * HEAD_DIM * SEQ + l;
      vb[(size_t)(nl)      * SEQ] = f2bf(a0[r]);
      vb[(size_t)(nl + 16) * SEQ] = f2bf(a1[r]);
      vb[(size_t)(nl + 32) * SEQ] = f2bf(a2[r]);
      vb[(size_t)(nl + 48) * SEQ] = f2bf(a3[r]);
    }
  }
}

// ---------------- kernel 3: flash attention ----------------
// One wave per (b, h, 16-query tile); streams 2048 keys in chunks of 32.
__global__ void __launch_bounds__(256)
flash_kernel(const bf16_t* __restrict__ qb, const bf16_t* __restrict__ kb,
             const bf16_t* __restrict__ vT, bf16_t* __restrict__ attn) {
  __shared__ __attribute__((aligned(32))) bf16_t plds[8][16][32]; // per-wave P staging (8 KB)

  const int lane = threadIdx.x & 31;
  const int slot = threadIdx.x >> 5;
  const int wid  = blockIdx.x * 8 + slot;
  const int qt   = wid & 127;          // L/16 = 128 query tiles
  const int h    = (wid >> 7) & 15;
  const int bi   = wid >> 11;
  const int kv   = h >> 2;             // repeat_interleave grouping
  const int nl   = lane & 15;
  const int g    = lane >> 4;
  const int row0 = qt * 16;

  // Q tile in A layout (16x64 bf16 -> two 16x32 operands)
  const bf16_t* qrow = qb + (((size_t)bi * NUM_HEADS + h) * SEQ + row0 + nl) * HEAD_DIM;
  const v16bf aq0 = load_a32(qrow, g);
  const v16bf aq1 = load_a32(qrow + 32, g);

  const bf16_t* kbase = kb + ((size_t)bi * NUM_KV + kv) * SEQ * HEAD_DIM;
  const bf16_t* vbase = vT + ((size_t)bi * NUM_KV + kv) * HEAD_DIM * SEQ;

  float mi[8], li[8];
#pragma unroll
  for (int r = 0; r < 8; r++) { mi[r] = -1e30f; li[r] = 0.0f; }
  v8f o0 = {}, o1 = {}, o2 = {}, o3 = {};

  for (int kbk = 0; kbk < SEQ; kbk += 32) {
    // ---- S = scale * Q K^T for 32 keys (two 16-key tiles) ----
    v8f s0 = {}, s1 = {};
    {
      const bf16_t* kr0 = kbase + (size_t)(kbk + nl) * HEAD_DIM;        // key = kbk+nl
      const bf16_t* kr1 = kbase + (size_t)(kbk + 16 + nl) * HEAD_DIM;   // key = kbk+16+nl
      v16bf bk00 = *(const v16bf*)(kr0 + 16 * g);       // hd 0..31
      v16bf bk01 = *(const v16bf*)(kr0 + 32 + 16 * g);  // hd 32..63
      v16bf bk10 = *(const v16bf*)(kr1 + 16 * g);
      v16bf bk11 = *(const v16bf*)(kr1 + 32 + 16 * g);
      s0 = wmma_bf16(aq0, bk00, s0);
      s0 = wmma_bf16(aq1, bk01, s0);
      s1 = wmma_bf16(aq0, bk10, s1);
      s1 = wmma_bf16(aq1, bk11, s1);
    }

    // ---- streaming softmax update (row = r + 8g, cols = keys across 16-lane half) ----
#pragma unroll
    for (int r = 0; r < 8; r++) {
      float v0 = s0[r] * ATTN_SCALE;
      float v1 = s1[r] * ATTN_SCALE;
      float mx = fmaxf(v0, v1);
      mx = fmaxf(mx, __shfl_xor(mx, 1));
      mx = fmaxf(mx, __shfl_xor(mx, 2));
      mx = fmaxf(mx, __shfl_xor(mx, 4));
      mx = fmaxf(mx, __shfl_xor(mx, 8));
      float mnew  = fmaxf(mi[r], mx);
      float alpha = __expf(mi[r] - mnew);
      float p0 = __expf(v0 - mnew);
      float p1 = __expf(v1 - mnew);
      float rs = p0 + p1;
      rs += __shfl_xor(rs, 1);
      rs += __shfl_xor(rs, 2);
      rs += __shfl_xor(rs, 4);
      rs += __shfl_xor(rs, 8);
      li[r] = li[r] * alpha + rs;
      mi[r] = mnew;
      o0[r] *= alpha; o1[r] *= alpha; o2[r] *= alpha; o3[r] *= alpha;
      // stage P (C layout -> [query row][key] in LDS)
      plds[slot][r + 8 * g][nl]      = f2bf(p0);
      plds[slot][r + 8 * g][16 + nl] = f2bf(p1);
    }
    asm volatile("s_wait_dscnt 0" ::: "memory");   // same-wave LDS RAW fence

    // read P back in A layout (16x32, K = 32 keys)
    v8bf plo = *(const v8bf*)&plds[slot][nl][16 * g];
    v8bf phi = *(const v8bf*)&plds[slot][nl][16 * g + 8];
    v16bf pa = cat8(plo, phi);

    // ---- O += P * V  (B from transposed V: contiguous along keys) ----
    const bf16_t* vr = vbase + kbk + 16 * g;
    v16bf bv0 = *(const v16bf*)(vr + (size_t)(nl)      * SEQ);
    v16bf bv1 = *(const v16bf*)(vr + (size_t)(nl + 16) * SEQ);
    v16bf bv2 = *(const v16bf*)(vr + (size_t)(nl + 32) * SEQ);
    v16bf bv3 = *(const v16bf*)(vr + (size_t)(nl + 48) * SEQ);
    o0 = wmma_bf16(pa, bv0, o0);
    o1 = wmma_bf16(pa, bv1, o1);
    o2 = wmma_bf16(pa, bv2, o2);
    o3 = wmma_bf16(pa, bv3, o3);
  }

  // ---- finalize: divide by row sums, store bf16 [B, L, H*64] ----
#pragma unroll
  for (int r = 0; r < 8; r++) {
    const float inv = 1.0f / li[r];
    const int row = row0 + r + 8 * g;
    bf16_t* orow = attn + ((size_t)bi * SEQ + row) * DIMX + h * HEAD_DIM;
    orow[nl]      = f2bf(o0[r] * inv);
    orow[nl + 16] = f2bf(o1[r] * inv);
    orow[nl + 32] = f2bf(o2[r] * inv);
    orow[nl + 48] = f2bf(o3[r] * inv);
  }
}

// ---------------- kernel 4: output projection ----------------
__global__ void __launch_bounds__(256)
oproj_kernel(const bf16_t* __restrict__ attn, const bf16_t* __restrict__ woT,
             float* __restrict__ out) {
  const int lane = threadIdx.x & 31;
  const int wid  = blockIdx.x * (blockDim.x >> 5) + (threadIdx.x >> 5);
  const int cg   = wid & 15;           // 16 column groups of 64
  const int rt   = wid >> 4;
  const int row0 = rt * 16;
  const int nl   = lane & 15;
  const int g    = lane >> 4;

  const bf16_t* arow = attn + (size_t)(row0 + nl) * DIMX;
  const bf16_t* base = woT + (size_t)(cg * 64 + nl) * DIMX + 16 * g;

  v8f a0 = {}, a1 = {}, a2 = {}, a3 = {};
  for (int kk = 0; kk < DIMX; kk += 32) {
    v16bf a = load_a32(arow + kk, g);
    __builtin_prefetch(arow + kk + 256, 0, 0);
    v16bf b0 = *(const v16bf*)(base + kk);
    v16bf b1 = *(const v16bf*)(base + (size_t)16 * DIMX + kk);
    v16bf b2 = *(const v16bf*)(base + (size_t)32 * DIMX + kk);
    v16bf b3 = *(const v16bf*)(base + (size_t)48 * DIMX + kk);
    a0 = wmma_bf16(a, b0, a0);
    a1 = wmma_bf16(a, b1, a1);
    a2 = wmma_bf16(a, b2, a2);
    a3 = wmma_bf16(a, b3, a3);
  }

#pragma unroll
  for (int r = 0; r < 8; r++) {
    const int row = row0 + r + 8 * g;
    float* orow = out + (size_t)row * DIMX + cg * 64;
    orow[nl]      = a0[r];
    orow[nl + 16] = a1[r];
    orow[nl + 32] = a2[r];
    orow[nl + 48] = a3[r];
  }
}

// ---------------- host launch ----------------
extern "C" void kernel_launch(void* const* d_in, const int* in_sizes, int n_in,
                              void* d_out, int out_size, void* d_ws, size_t ws_size,
                              hipStream_t stream) {
  const float* x  = (const float*)d_in[0];
  const float* Wq = (const float*)d_in[1];
  const float* Wk = (const float*)d_in[2];
  const float* Wv = (const float*)d_in[3];
  const float* Wo = (const float*)d_in[4];
  float* out = (float*)d_out;

  // workspace carve-up (bytes)
  char* ws = (char*)d_ws;
  bf16_t* xb   = (bf16_t*)(ws);                      // 4096*1024 bf16 = 8 MiB
  bf16_t* wqT  = (bf16_t*)(ws + (8u << 20));         // 2 MiB
  bf16_t* wkT  = (bf16_t*)(ws + (10u << 20));        // 0.5 MiB
  bf16_t* wvT  = (bf16_t*)(ws + (10u << 20) + (512u << 10));
  bf16_t* woT  = (bf16_t*)(ws + (11u << 20));        // 2 MiB
  bf16_t* qb   = (bf16_t*)(ws + (13u << 20));        // 8 MiB
  bf16_t* kbuf = (bf16_t*)(ws + (21u << 20));        // 2 MiB
  bf16_t* vT   = (bf16_t*)(ws + (23u << 20));        // 2 MiB
  bf16_t* attn = (bf16_t*)(ws + (25u << 20));        // 8 MiB  (total 33 MiB)

  const int nx = NROWS * DIMX; // 4,194,304
  cvt_bf16_kernel<<<nx / 256, 256, 0, stream>>>(x, xb, nx);
  transpose_cvt_kernel<<<(DIMX * DIMX) / 256, 256, 0, stream>>>(Wq, wqT, DIMX, DIMX);
  transpose_cvt_kernel<<<(DIMX * 256) / 256, 256, 0, stream>>>(Wk, wkT, DIMX, 256);
  transpose_cvt_kernel<<<(DIMX * 256) / 256, 256, 0, stream>>>(Wv, wvT, DIMX, 256);
  transpose_cvt_kernel<<<(DIMX * DIMX) / 256, 256, 0, stream>>>(Wo, woT, DIMX, DIMX);

  // QKV+RoPE: (NROWS/16) row tiles * 24 col groups = 6144 waves -> 768 blocks of 8 waves
  qkv_rope_kernel<<<768, 256, 0, stream>>>(xb, wqT, wkT, wvT, qb, kbuf, vT);

  // flash attention: B*H*(L/16) = 4096 waves -> 512 blocks
  flash_kernel<<<512, 256, 0, stream>>>(qb, kbuf, vT, attn);

  // output projection: 256 row tiles * 16 col groups = 4096 waves -> 512 blocks
  oproj_kernel<<<512, 256, 0, stream>>>(attn, woT, out);
}
